// Att_BiLSTM_CRF_78889959293357
// MI455X (gfx1250) — compile-verified, matrix operand order checked by reading, not run on hardware
//
#include <hip/hip_runtime.h>
#include <hip/hip_bf16.h>

// ---------------------------------------------------------------------------
// Att-BiLSTM-CRF for MI455X (gfx1250, wave32, WMMA).
//   B=32, S=512, D=H1=H2=256, NS=16, T=12, START=10, STOP=11
// Pipeline:
//   1) bf16 weight packs + embedding gather (bf16 activations for WMMA)
//   2) big WMMA GEMMs for time-independent input projections XP = X @ Wih^T
//      (A tile staged block-wide into LDS via async global->LDS DMA)
//   3) sequential LSTM recurrence: Whh resident in VGPR WMMA fragments,
//      h passes through LDS each step, f32 gates in dynamic LDS
//   4) attention (wx = L1 @ attW^T via WMMA; softmax over 16 per row in LDS)
//   5) second BiLSTM (same recurrence kernel), 12-tag head, Viterbi.
// ---------------------------------------------------------------------------

typedef __bf16 bf16_t;
typedef __attribute__((ext_vector_type(16))) __bf16 v16bf;
typedef __attribute__((ext_vector_type(8)))  float  v8f;

#define NB   32      // batch
#define NSEQ 512     // sequence length
#define NT   12      // tags
#define TAG_START 10
#define TAG_STOP  11

__device__ __forceinline__ float sigf(float x) { return 1.0f / (1.0f + __expf(-x)); }

// Load a 16x32 bf16 WMMA operand fragment (A or B) from a row-major matrix.
// ISA 7.12.2: lane&15 selects row (A: M) / col (B: N); per-VGPR K mapping is
// K = k0 + 8*(lane>=16) + (v<4 ? 2v : 2v+8).
__device__ __forceinline__ v16bf load_frag(const bf16_t* __restrict__ base,
                                           int ld, int r, int k0, int kb) {
  v16bf x;
#pragma unroll
  for (int v = 0; v < 8; ++v) {
    int k = k0 + kb + ((v < 4) ? (2 * v) : (2 * v + 8));
    x[2 * v]     = base[(size_t)r * ld + k];
    x[2 * v + 1] = base[(size_t)r * ld + k + 1];
  }
  return x;
}

// ---------------------------------------------------------------------------
// f32 -> bf16 conversion (weights / activations)
// ---------------------------------------------------------------------------
__global__ void cvt_bf16_kernel(const float* __restrict__ src,
                                bf16_t* __restrict__ dst, long n) {
  long i = (long)blockIdx.x * blockDim.x + threadIdx.x;
  long stride = (long)gridDim.x * blockDim.x;
  for (; i < n; i += stride) dst[i] = (bf16_t)src[i];
}

// ---------------------------------------------------------------------------
// Embedding gather: X1[(s*32+b), d] = embed[inputs[b,s], d]  (bf16)
// ---------------------------------------------------------------------------
__global__ void gather_kernel(const int* __restrict__ tok,
                              const float* __restrict__ E,
                              bf16_t* __restrict__ X) {
  int idx = blockIdx.x * blockDim.x + threadIdx.x;
  if (idx >= NSEQ * NB * 256) return;
  int d = idx & 255;
  int row = idx >> 8;          // s*32 + b
  int b = row & 31;
  int s = row >> 5;
  int t = tok[b * NSEQ + s];
  X[idx] = (bf16_t)E[(size_t)t * 256 + d];
}

// ---------------------------------------------------------------------------
// Generic WMMA GEMM: C[M,N] (f32) = A[M,K] (bf16, row-major) @ W[N,K]^T (bf16)
// One wave computes a 16x64 tile (4 accumulators); 8 waves / block.
// All 8 waves of a block share the same 16-row A tile (ng % 8 == 0), so the
// contiguous 16xK A slab is staged once into LDS with async global->LDS
// copies (ASYNCcnt), then A fragments are read via DS.
// Requires N % 64 == 0, M % 16 == 0, K % 32 == 0, grid = M/16 * N/64 / 8.
// Dynamic LDS: 16*K*2 bytes.
// ---------------------------------------------------------------------------
__global__ void __launch_bounds__(256)
gemm_bf16_nt(const bf16_t* __restrict__ A, const bf16_t* __restrict__ W,
             float* __restrict__ C, int M, int N, int K) {
  extern __shared__ char smem_raw[];
  bf16_t* As = (bf16_t*)smem_raw;                  // 16 x K bf16
  int wave = threadIdx.x >> 5;
  int lane = threadIdx.x & 31;
  int ng = N >> 6;
  int task = blockIdx.x * 8 + wave;
  int mt = task / ng;
  int nb = task - mt * ng;
  int row0 = mt << 4, col0 = nb << 6;
  int kb = (lane >> 4) << 3;

  // ---- stage A tile (contiguous 16*K*2 bytes) into LDS asynchronously ----
  {
    const int total = 16 * K * 2;                  // bytes
    unsigned lds0 = (unsigned)(size_t)As;          // LDS byte offset
    const char* g0 = (const char*)(A + (size_t)row0 * K);
    for (int off = threadIdx.x * 16; off < total; off += 256 * 16) {
      unsigned l = lds0 + (unsigned)off;
      const void* g = (const void*)(g0 + off);
      asm volatile("global_load_async_to_lds_b128 %0, %1, off"
                   :: "v"(l), "v"(g) : "memory");
    }
    asm volatile("s_wait_asynccnt 0" ::: "memory");
  }
  __syncthreads();

  v8f acc[4];
#pragma unroll
  for (int i = 0; i < 4; ++i)
#pragma unroll
    for (int r = 0; r < 8; ++r) acc[i][r] = 0.0f;

  for (int k0 = 0; k0 < K; k0 += 32) {
    v16bf a = load_frag(As, K, (lane & 15), k0, kb);   // from LDS
#pragma unroll
    for (int nt = 0; nt < 4; ++nt) {
      int n = col0 + (nt << 4) + (lane & 15);
      v16bf b = load_frag(W, K, n, k0, kb);            // streamed from global
      acc[nt] = __builtin_amdgcn_wmma_f32_16x16x32_bf16(
          false, a, false, b, (short)0, acc[nt], false, false);
    }
  }
  int mlo = row0 + ((lane >> 4) << 3);
#pragma unroll
  for (int nt = 0; nt < 4; ++nt) {
    int n = col0 + (nt << 4) + (lane & 15);
#pragma unroll
    for (int r = 0; r < 8; ++r) C[(size_t)(mlo + r) * N + n] = acc[nt][r];
  }
}

// ---------------------------------------------------------------------------
// LSTM recurrence (one direction per block; grid = 2).
//   gates(32x1024) = XP[t] + h(32x256) @ Whh^T(1024x256) + bias
// 32 waves: wave w owns output cols [32w, 32w+32), keeps its Whh fragments
// resident in VGPRs (2 n-tiles x 8 k-chunks x 8 VGPRs = 128 VGPRs) for all
// 512 steps. Dynamic LDS: gates 128KB + h(bf16) 16KB + c(f32) 32KB = 176KB.
// ---------------------------------------------------------------------------
__global__ void __launch_bounds__(1024)
lstm_recur(const bf16_t* __restrict__ Whh,   // (2, 1024, 256) bf16
           const float* __restrict__ bias,   // (2, 1024)
           const float* __restrict__ XP,     // (2, S*B, 1024) f32
           float* __restrict__ OUT) {        // (S*B, 512), dir -> col offset
  extern __shared__ char smem[];
  float*  gg = (float*)smem;                                   // 32*1024 f32
  bf16_t* hb = (bf16_t*)(smem + 32 * 1024 * sizeof(float));    // 32*256 bf16
  float*  cc = (float*)(smem + 32 * 1024 * sizeof(float)
                             + 32 * 256 * sizeof(bf16_t));     // 32*256 f32
  const int dir  = blockIdx.x;
  const bf16_t* W    = Whh  + (size_t)dir * 1024 * 256;
  const float*  bptr = bias + (size_t)dir * 1024;
  const float*  xp   = XP   + (size_t)dir * NSEQ * NB * 1024;
  const int tid = threadIdx.x, wave = tid >> 5, lane = tid & 31;
  const int n0 = wave << 5;
  const int kb = (lane >> 4) << 3;

  // Resident Whh B-fragments.
  v16bf bfrag[2][8];
#pragma unroll
  for (int nt = 0; nt < 2; ++nt) {
    int n = n0 + (nt << 4) + (lane & 15);
#pragma unroll
    for (int kc = 0; kc < 8; ++kc)
      bfrag[nt][kc] = load_frag(W, 256, n, kc << 5, kb);
  }

  // Activation-phase assignment: thread -> (batch ab, 8 cols from j0).
  const int ab = wave;
  const int j0 = lane << 3;
  float bi[8], bff[8], bgg[8], boo[8];
#pragma unroll
  for (int u = 0; u < 8; ++u) {
    bi[u]  = bptr[j0 + u];
    bff[u] = bptr[256 + j0 + u];
    bgg[u] = bptr[512 + j0 + u];
    boo[u] = bptr[768 + j0 + u];
  }

  for (int e = tid; e < 32 * 256; e += 1024) { hb[e] = (bf16_t)0.0f; cc[e] = 0.0f; }
  __syncthreads();

  for (int step = 0; step < NSEQ; ++step) {
    const int s = dir ? (NSEQ - 1 - step) : step;

    // Prefetch next step's XP row while the gate GEMM runs.
    if (step + 1 < NSEQ) {
      const int s2 = dir ? (NSEQ - 2 - step) : (step + 1);
      __builtin_prefetch(xp + ((size_t)(s2 * NB + ab)) * 1024 + j0, 0, 0);
      __builtin_prefetch(xp + ((size_t)(s2 * NB + ab)) * 1024 + 512 + j0, 0, 0);
    }

    // ---- gate GEMM: h @ Whh^T ----
    v8f acc[2][2];
#pragma unroll
    for (int mt = 0; mt < 2; ++mt)
#pragma unroll
      for (int nt = 0; nt < 2; ++nt)
#pragma unroll
        for (int r = 0; r < 8; ++r) acc[mt][nt][r] = 0.0f;

#pragma unroll
    for (int kc = 0; kc < 8; ++kc) {
      v16bf a0 = load_frag(hb, 256, (lane & 15),      kc << 5, kb);
      v16bf a1 = load_frag(hb, 256, 16 + (lane & 15), kc << 5, kb);
      acc[0][0] = __builtin_amdgcn_wmma_f32_16x16x32_bf16(
          false, a0, false, bfrag[0][kc], (short)0, acc[0][0], false, false);
      acc[0][1] = __builtin_amdgcn_wmma_f32_16x16x32_bf16(
          false, a0, false, bfrag[1][kc], (short)0, acc[0][1], false, false);
      acc[1][0] = __builtin_amdgcn_wmma_f32_16x16x32_bf16(
          false, a1, false, bfrag[0][kc], (short)0, acc[1][0], false, false);
      acc[1][1] = __builtin_amdgcn_wmma_f32_16x16x32_bf16(
          false, a1, false, bfrag[1][kc], (short)0, acc[1][1], false, false);
    }
#pragma unroll
    for (int mt = 0; mt < 2; ++mt) {
      int mlo = (mt << 4) + ((lane >> 4) << 3);
#pragma unroll
      for (int nt = 0; nt < 2; ++nt) {
        int n = n0 + (nt << 4) + (lane & 15);
#pragma unroll
        for (int r = 0; r < 8; ++r) gg[(mlo + r) * 1024 + n] = acc[mt][nt][r];
      }
    }
    __syncthreads();

    // ---- activations + state update ----
    {
      const float* xr = xp + ((size_t)(s * NB + ab)) * 1024;
      float* outr = OUT + ((size_t)(s * NB + ab)) * 512 + dir * 256;
#pragma unroll
      for (int u = 0; u < 8; ++u) {
        int j = j0 + u;
        float iv = gg[ab * 1024 + j]       + xr[j]       + bi[u];
        float fv = gg[ab * 1024 + 256 + j] + xr[256 + j] + bff[u];
        float gv = gg[ab * 1024 + 512 + j] + xr[512 + j] + bgg[u];
        float ov = gg[ab * 1024 + 768 + j] + xr[768 + j] + boo[u];
        float cn = sigf(fv) * cc[ab * 256 + j] + sigf(iv) * tanhf(gv);
        float hn = sigf(ov) * tanhf(cn);
        cc[ab * 256 + j] = cn;
        hb[ab * 256 + j] = (bf16_t)hn;
        outr[j] = hn;
      }
    }
    __syncthreads();
  }
}

// ---------------------------------------------------------------------------
// Attention per row (s,b): scores over NS=16 sent embeddings, softmax,
// context g, emit comb = [wx | g] as bf16. grid = S*B, block = 64.
// ---------------------------------------------------------------------------
__global__ void __launch_bounds__(64)
attn_kernel(const float* __restrict__ WX,     // (S*B, 512)
            const float* __restrict__ sent,   // (B, 16, 512)
            bf16_t* __restrict__ COMB) {      // (S*B, 1024)
  __shared__ float wxr[512];
  __shared__ float sc[16];
  int row = blockIdx.x;   // s*32 + b
  int b = row & 31;
  int tid = threadIdx.x;
  const float* wrow = WX + (size_t)row * 512;
  for (int f = tid; f < 512; f += 64) wxr[f] = wrow[f];
  __syncthreads();
  if (tid < 16) {
    const float* sr = sent + ((size_t)b * 16 + tid) * 512;
    float d = 0.0f;
    for (int f = 0; f < 512; ++f) d += wxr[f] * sr[f];
    sc[tid] = d;
  }
  __syncthreads();
  if (tid == 0) {
    float mx = sc[0];
    for (int n = 1; n < 16; ++n) mx = fmaxf(mx, sc[n]);
    float ss = 0.0f;
    for (int n = 0; n < 16; ++n) { sc[n] = __expf(sc[n] - mx); ss += sc[n]; }
    float inv = 1.0f / ss;
    for (int n = 0; n < 16; ++n) sc[n] *= inv;
  }
  __syncthreads();
  bf16_t* crow = COMB + (size_t)row * 1024;
  for (int f = tid; f < 512; f += 64) {
    float g = 0.0f;
#pragma unroll
    for (int n = 0; n < 16; ++n) g += sc[n] * sent[((size_t)b * 16 + n) * 512 + f];
    crow[f]       = (bf16_t)wxr[f];
    crow[512 + f] = (bf16_t)g;
  }
}

// ---------------------------------------------------------------------------
// Tag head: feats[b,s,t] = L2[(s*32+b)] . h2t_w[t] + h2t_b[t]
// ---------------------------------------------------------------------------
__global__ void feats_kernel(const float* __restrict__ L2,
                             const float* __restrict__ hw,
                             const float* __restrict__ hb,
                             float* __restrict__ F) {
  int idx = blockIdx.x * blockDim.x + threadIdx.x;
  if (idx >= NB * NSEQ * NT) return;
  int t = idx % NT;
  int s = (idx / NT) & (NSEQ - 1);
  int b = idx / (NT * NSEQ);
  const float* r = L2 + ((size_t)(s * NB + b)) * 512;
  const float* w = hw + t * 512;
  float d = hb[t];
  for (int f = 0; f < 512; ++f) d += r[f] * w[f];
  F[((size_t)b * NSEQ + s) * NT + t] = d;
}

// ---------------------------------------------------------------------------
// Viterbi per batch (grid = 32, block = 32; one wave).
// ---------------------------------------------------------------------------
__global__ void viterbi_kernel(const float* __restrict__ F,
                               const float* __restrict__ trans,
                               int* __restrict__ BP,
                               float* __restrict__ OUT) {
  __shared__ float fv[NT], nfv[NT];
  int b = blockIdx.x, tid = threadIdx.x;
  const float* fb = F + (size_t)b * NSEQ * NT;
  int* bp = BP + (size_t)b * NSEQ * NT;
  if (tid < NT) fv[tid] = (tid == TAG_START) ? 0.0f : -10000.0f;
  __syncthreads();
  for (int s = 0; s < NSEQ; ++s) {
    if (tid < NT) {
      float best = -3.4e38f; int bi = 0;
#pragma unroll
      for (int i = 0; i < NT; ++i) {
        float v = fv[i] + trans[tid * NT + i];
        if (v > best) { best = v; bi = i; }
      }
      nfv[tid] = best + fb[s * NT + tid];
      bp[s * NT + tid] = bi;
    }
    __syncthreads();
    if (tid < NT) fv[tid] = nfv[tid];
    __syncthreads();
  }
  if (tid == 0) {
    float best = -3.4e38f; int bt = 0;
    for (int j = 0; j < NT; ++j) {
      float v = fv[j] + trans[TAG_STOP * NT + j];
      if (v > best) { best = v; bt = j; }
    }
    int tag = bt;
    for (int s = NSEQ - 1; s >= 0; --s) {
      OUT[(size_t)b * NSEQ + s] = (float)tag;
      tag = bp[s * NT + tag];
    }
  }
}

// ---------------------------------------------------------------------------
// Host launch
// ---------------------------------------------------------------------------
extern "C" void kernel_launch(void* const* d_in, const int* in_sizes, int n_in,
                              void* d_out, int out_size, void* d_ws, size_t ws_size,
                              hipStream_t stream) {
  (void)in_sizes; (void)n_in; (void)out_size; (void)ws_size;
  const int*   inputs = (const int*)d_in[0];
  const float* sent   = (const float*)d_in[1];
  const float* embed  = (const float*)d_in[2];
  const float* w_ih1  = (const float*)d_in[3];
  const float* w_hh1  = (const float*)d_in[4];
  const float* b1     = (const float*)d_in[5];
  const float* w_ih2  = (const float*)d_in[6];
  const float* w_hh2  = (const float*)d_in[7];
  const float* b2     = (const float*)d_in[8];
  const float* attW   = (const float*)d_in[9];
  const float* h2t_w  = (const float*)d_in[10];
  const float* h2t_b  = (const float*)d_in[11];
  const float* trans  = (const float*)d_in[12];
  float* out = (float*)d_out;
  char* ws = (char*)d_ws;

  size_t o = 0;
  auto alloc = [&](size_t bytes) { size_t p = o; o += (bytes + 255) & ~(size_t)255; return p; };
  const size_t oWIH1 = alloc((size_t)2 * 1024 * 256 * 2);
  const size_t oWHH1 = alloc((size_t)2 * 1024 * 256 * 2);
  const size_t oWIH2 = alloc((size_t)2 * 1024 * 1024 * 2);
  const size_t oWHH2 = alloc((size_t)2 * 1024 * 256 * 2);
  const size_t oATTW = alloc((size_t)512 * 512 * 2);
  const size_t oX1   = alloc((size_t)NSEQ * NB * 256 * 2);
  const size_t oXP   = alloc((size_t)2 * NSEQ * NB * 1024 * 4);
  const size_t oL1   = alloc((size_t)NSEQ * NB * 512 * 4);
  const size_t oL1B  = alloc((size_t)NSEQ * NB * 512 * 2);
  const size_t oWX   = alloc((size_t)NSEQ * NB * 512 * 4);
  const size_t oCOMB = alloc((size_t)NSEQ * NB * 1024 * 2);
  const size_t oL2   = alloc((size_t)NSEQ * NB * 512 * 4);
  const size_t oF    = alloc((size_t)NB * NSEQ * NT * 4);
  const size_t oBP   = alloc((size_t)NB * NSEQ * NT * 4);

  bf16_t* WIH1 = (bf16_t*)(ws + oWIH1);
  bf16_t* WHH1 = (bf16_t*)(ws + oWHH1);
  bf16_t* WIH2 = (bf16_t*)(ws + oWIH2);
  bf16_t* WHH2 = (bf16_t*)(ws + oWHH2);
  bf16_t* ATTB = (bf16_t*)(ws + oATTW);
  bf16_t* X1   = (bf16_t*)(ws + oX1);
  float*  XPf  = (float*)(ws + oXP);
  float*  L1f  = (float*)(ws + oL1);
  bf16_t* L1B  = (bf16_t*)(ws + oL1B);
  float*  WXf  = (float*)(ws + oWX);
  bf16_t* COMB = (bf16_t*)(ws + oCOMB);
  float*  L2f  = (float*)(ws + oL2);
  float*  Ff   = (float*)(ws + oF);
  int*    BPi  = (int*)(ws + oBP);

  // 1) pack weights to bf16 + embedding gather
  cvt_bf16_kernel<<<1024, 256, 0, stream>>>(w_ih1, WIH1, (long)2 * 1024 * 256);
  cvt_bf16_kernel<<<1024, 256, 0, stream>>>(w_hh1, WHH1, (long)2 * 1024 * 256);
  cvt_bf16_kernel<<<2048, 256, 0, stream>>>(w_ih2, WIH2, (long)2 * 1024 * 1024);
  cvt_bf16_kernel<<<1024, 256, 0, stream>>>(w_hh2, WHH2, (long)2 * 1024 * 256);
  cvt_bf16_kernel<<<512, 256, 0, stream>>>(attW, ATTB, (long)512 * 512);
  gather_kernel<<<(NSEQ * NB * 256) / 256, 256, 0, stream>>>(inputs, embed, X1);

  // 2) layer-1 input projections (WMMA, chip-wide); LDS A-stage = 16*K*2 B
  for (int dir = 0; dir < 2; ++dir)
    gemm_bf16_nt<<<2048, 256, 16 * 256 * 2, stream>>>(
        X1, WIH1 + (size_t)dir * 1024 * 256,
        XPf + (size_t)dir * NSEQ * NB * 1024, NSEQ * NB, 1024, 256);

  // 3) layer-1 recurrence (fwd + bwd blocks)
  const size_t shmem = 32 * 1024 * sizeof(float) + 32 * 256 * sizeof(bf16_t)
                     + 32 * 256 * sizeof(float);
  lstm_recur<<<2, 1024, shmem, stream>>>(WHH1, b1, XPf, L1f);

  // 4) attention
  cvt_bf16_kernel<<<4096, 256, 0, stream>>>(L1f, L1B, (long)NSEQ * NB * 512);
  gemm_bf16_nt<<<1024, 256, 16 * 512 * 2, stream>>>(L1B, ATTB, WXf,
                                                    NSEQ * NB, 512, 512);
  attn_kernel<<<NSEQ * NB, 64, 0, stream>>>(WXf, sent, COMB);

  // 5) layer-2 input projections + recurrence
  for (int dir = 0; dir < 2; ++dir)
    gemm_bf16_nt<<<2048, 256, 16 * 1024 * 2, stream>>>(
        COMB, WIH2 + (size_t)dir * 1024 * 1024,
        XPf + (size_t)dir * NSEQ * NB * 1024, NSEQ * NB, 1024, 1024);
  lstm_recur<<<2, 1024, shmem, stream>>>(WHH2, b2, XPf, L2f);

  // 6) tag head + Viterbi
  feats_kernel<<<(NB * NSEQ * NT + 255) / 256, 256, 0, stream>>>(L2f, h2t_w, h2t_b, Ff);
  viterbi_kernel<<<NB, 32, 0, stream>>>(Ff, trans, BPi, out);
}
